// QGATMultiHeadLayer_88905823027443
// MI455X (gfx1250) — matrix operand order repaired and checked by reading, not compiled, not run on hardware
//
#include <hip/hip_runtime.h>
#include <hip/hip_bf16.h>

#define ALPHA   0.2f
#define NEG_INF -9000000000000000.0f

typedef __attribute__((ext_vector_type(16))) __bf16         v16bf;
typedef __attribute__((ext_vector_type(8)))  float          v8f;
typedef __attribute__((ext_vector_type(4)))  float          v4f;
typedef __attribute__((ext_vector_type(4)))  int            v4i;
typedef __attribute__((ext_vector_type(8)))  unsigned short v8u16;
typedef __attribute__((ext_vector_type(4)))  unsigned short v4u16;

// 32-byte fragment: two 16B halves viewed as a v16bf WMMA operand
union Frag32B { struct { v8u16 lo, hi; } s; v16bf v; };

static __device__ __forceinline__ unsigned short f32_bf16(float f) {
    unsigned int u = __float_as_uint(f);
    u += 0x7FFFu + ((u >> 16) & 1u);      // round-to-nearest-even
    return (unsigned short)(u >> 16);
}

// ---------------- prep: x (f32) -> bf16 ----------------
__global__ void k_x2bf(const float* __restrict__ x, unsigned short* __restrict__ xb, int n) {
    for (int i = blockIdx.x * blockDim.x + threadIdx.x; i < n; i += gridDim.x * blockDim.x)
        xb[i] = f32_bf16(x[i]);
}

// ---------------- prep: hamiltonT (H,128,512) bf16 from W (H,128,128) ----------------
__global__ void k_ham(const float* __restrict__ W, unsigned short* __restrict__ hamT) {
    int idx = blockIdx.x * blockDim.x + threadIdx.x;     // over H*128*512
    if (idx >= 4 * 128 * 512) return;
    int head = idx / (128 * 512);
    int rem  = idx - head * (128 * 512);
    int o = rem >> 9;              // 0..127  (output col of hamilton)
    int f = rem & 511;             // 0..511  (input row of hamilton)
    int qc = o >> 5, oc = o & 31;
    int qr = f >> 7, fr = f & 127;
    const int   comp[4][4] = {{0,1,2,3},{1,0,3,2},{2,3,0,1},{3,2,1,0}};
    const float sgn [4][4] = {{ 1, 1, 1, 1},{-1, 1, 1,-1},{-1,-1, 1, 1},{-1, 1,-1, 1}};
    float v = sgn[qr][qc] * W[(head * 128 + fr) * 128 + comp[qr][qc] * 32 + oc];
    hamT[idx] = f32_bf16(v);       // hamT[head][o][f]
}

// ---------------- h = x @ hamilton : (H,4096,128) via bf16 WMMA ----------------
// block = 256 threads (8 waves); block tile 128(M) x 64(N); wave tile 32x32 (2x2)
__global__ void k_gemm_h(const unsigned short* __restrict__ xb,   // (4096,512) bf16
                         const unsigned short* __restrict__ hamT, // (H,128,512) bf16
                         float* __restrict__ hf,                  // (H,4096,128) f32
                         unsigned short* __restrict__ hTb)        // (H,128,4096) bf16
{
    const int head  = blockIdx.z;
    const int n_blk = blockIdx.y * 64;
    const int m_blk = blockIdx.x * 128;
    const int tid  = threadIdx.x;
    const int wave = tid >> 5, lane = tid & 31;
    const int half = lane >> 4, l16 = lane & 15;
    const int wm = wave >> 1;            // 0..3 -> 32 rows each
    const int wn = wave & 1;             // 0..1 -> 32 cols each

    v8f acc[2][2];
    #pragma unroll
    for (int i = 0; i < 2; ++i)
        #pragma unroll
        for (int j = 0; j < 2; ++j) acc[i][j] = v8f{};

    const unsigned short* hamH = hamT + head * (128 * 512);

    for (int kk = 0; kk < 512; kk += 32) {
        Frag32B a[2], b[2];
        #pragma unroll
        for (int i = 0; i < 2; ++i) {
            int row = m_blk + wm * 32 + i * 16 + l16;
            const unsigned short* pa = xb + row * 512 + kk + half * 8;
            a[i].s.lo = *(const v8u16*)(pa);
            a[i].s.hi = *(const v8u16*)(pa + 16);
        }
        #pragma unroll
        for (int j = 0; j < 2; ++j) {
            int o = n_blk + wn * 32 + j * 16 + l16;
            const unsigned short* pb = hamH + o * 512 + kk + half * 8;
            b[j].s.lo = *(const v8u16*)(pb);
            b[j].s.hi = *(const v8u16*)(pb + 16);
        }
        #pragma unroll
        for (int i = 0; i < 2; ++i)
            #pragma unroll
            for (int j = 0; j < 2; ++j)
                acc[i][j] = __builtin_amdgcn_wmma_f32_16x16x32_bf16(
                    false, a[i].v, false, b[j].v, (short)0, acc[i][j], false, false);
    }

    #pragma unroll
    for (int i = 0; i < 2; ++i)
        #pragma unroll
        for (int j = 0; j < 2; ++j) {
            int o = n_blk + wn * 32 + j * 16 + l16;
            #pragma unroll
            for (int r = 0; r < 8; ++r) {
                int n = m_blk + wm * 32 + i * 16 + half * 8 + r;
                float v = acc[i][j][r];
                hf[(head * 4096 + n) * 128 + o]  = v;
                hTb[(head * 128 + o) * 4096 + n] = f32_bf16(v);
            }
        }
}

// ---------------- f_src/f_dst = h . a ----------------
__global__ void k_logits(const float* __restrict__ hf, const float* __restrict__ a,
                         float* __restrict__ fsrc, float* __restrict__ fdst)
{
    int idx = blockIdx.x * blockDim.x + threadIdx.x;   // h*4096 + n
    if (idx >= 4 * 4096) return;
    int head = idx >> 12;
    const float* hr = hf + idx * 128;
    const float* as = a + head * 256;
    const float* ad = as + 128;
    float s0 = 0.f, s1 = 0.f;
    #pragma unroll 4
    for (int o = 0; o < 128; ++o) { float hv = hr[o]; s0 += hv * as[o]; s1 += hv * ad[o]; }
    fsrc[idx] = s0; fdst[idx] = s1;
}

// ---------------- fused masked softmax + attn@h + ELU ----------------
// block: 256 threads, handles (head, 16 rows). LDS: P tile 16x4096 bf16.
__global__ void k_attn(const int* __restrict__ adj,
                       const float* __restrict__ fsrc,
                       const float* __restrict__ fdst,
                       const unsigned short* __restrict__ hTb,   // (H,128,4096) bf16
                       float* __restrict__ out)                  // (4096,512) f32
{
    extern __shared__ char smem[];
    unsigned short* p  = (unsigned short*)smem;                  // 16*4096 bf16
    float* red    = (float*)(smem + 16 * 4096 * 2);              // 256 f32
    float* rowinv = red + 256;                                   // 16 f32

    const int head = blockIdx.y;
    const int n0   = blockIdx.x * 16;
    const int tid  = threadIdx.x;
    const int row  = tid >> 4;          // 0..15
    const int c16  = tid & 15;

    const float  fs   = fsrc[head * 4096 + n0 + row];
    const float* fd   = fdst + head * 4096;
    const int*   arow = adj + (n0 + row) * 4096;

    // pass 1: masked row max (vectorized: 4 neighbors / iter / thread)
    float lmax = -INFINITY;
    for (int m = c16 * 4; m < 4096; m += 64) {
        v4i av = *(const v4i*)(arow + m);
        v4f fv = *(const v4f*)(fd + m);
        #pragma unroll
        for (int q = 0; q < 4; ++q) {
            float e = fs + fv[q];
            e = e > 0.0f ? e : ALPHA * e;
            e = (av[q] > 0) ? e : NEG_INF;
            lmax = fmaxf(lmax, e);
        }
    }
    red[tid] = lmax;
    __syncthreads();
    float rmax = red[row * 16];
    #pragma unroll
    for (int i = 1; i < 16; ++i) rmax = fmaxf(rmax, red[row * 16 + i]);
    __syncthreads();

    // pass 2: unnormalized p = exp(e - max) -> LDS (bf16); row sum
    float lsum = 0.0f;
    for (int m = c16 * 4; m < 4096; m += 64) {
        v4i av = *(const v4i*)(arow + m);
        v4f fv = *(const v4f*)(fd + m);
        v4u16 pv4;
        #pragma unroll
        for (int q = 0; q < 4; ++q) {
            float e = fs + fv[q];
            e = e > 0.0f ? e : ALPHA * e;
            e = (av[q] > 0) ? e : NEG_INF;
            float pv = __expf(e - rmax);
            pv4[q] = f32_bf16(pv);
            lsum += pv;
        }
        *(v4u16*)(p + row * 4096 + m) = pv4;
    }
    red[tid] = lsum;
    __syncthreads();
    if (c16 == 0) {
        float s = 0.0f;
        #pragma unroll
        for (int i = 0; i < 16; ++i) s += red[row * 16 + i];
        rowinv[row] = 1.0f / s;
    }
    __syncthreads();

    // P(16x4096) @ V(4096x128): each wave owns 16 output cols, K=4096
    // Two independent accumulator chains (K split in halves) for WMMA ILP.
    const int wave = tid >> 5, lane = tid & 31;
    const int half = lane >> 4, l16 = lane & 15;
    const int o = wave * 16 + l16;
    const unsigned short* vrow = hTb + (head * 128 + o) * 4096;
    const unsigned short* prow = p + l16 * 4096;

    v8f acc0 = v8f{}, acc1 = v8f{};
    for (int kk = 0; kk < 2048; kk += 32) {
        Frag32B a0, b0, a1, b1;
        const unsigned short* pa0 = prow + kk + half * 8;          // LDS
        const unsigned short* pa1 = pa0 + 2048;
        a0.s.lo = *(const v8u16*)(pa0);
        a0.s.hi = *(const v8u16*)(pa0 + 16);
        a1.s.lo = *(const v8u16*)(pa1);
        a1.s.hi = *(const v8u16*)(pa1 + 16);
        const unsigned short* pb0 = vrow + kk + half * 8;          // global (L2-resident)
        const unsigned short* pb1 = pb0 + 2048;
        b0.s.lo = *(const v8u16*)(pb0);
        b0.s.hi = *(const v8u16*)(pb0 + 16);
        b1.s.lo = *(const v8u16*)(pb1);
        b1.s.hi = *(const v8u16*)(pb1 + 16);
        acc0 = __builtin_amdgcn_wmma_f32_16x16x32_bf16(
            false, a0.v, false, b0.v, (short)0, acc0, false, false);
        acc1 = __builtin_amdgcn_wmma_f32_16x16x32_bf16(
            false, a1.v, false, b1.v, (short)0, acc1, false, false);
    }

    // epilogue: merge chains, normalize by row sum, ELU, scatter
    #pragma unroll
    for (int r = 0; r < 8; ++r) {
        int rr = r + half * 8;
        float v = (acc0[r] + acc1[r]) * rowinv[rr];
        v = v > 0.0f ? v : expm1f(v);
        out[(n0 + rr) * 512 + head * 128 + o] = v;
    }
}

extern "C" void kernel_launch(void* const* d_in, const int* in_sizes, int n_in,
                              void* d_out, int out_size, void* d_ws, size_t ws_size,
                              hipStream_t stream) {
    const float* x   = (const float*)d_in[0];   // (4096,512)
    const int*   adj = (const int*)d_in[1];     // (4096,4096)
    const float* W   = (const float*)d_in[2];   // (4,128,128)
    const float* a   = (const float*)d_in[3];   // (4,256)
    float* out = (float*)d_out;                 // (4096,512)

    char* ws = (char*)d_ws;
    unsigned short* xb   = (unsigned short*)(ws);              //  4,194,304 B
    unsigned short* hamT = (unsigned short*)(ws +  4194304);   //    524,288 B
    float*          hf   = (float*)         (ws +  4718592);   //  8,388,608 B
    unsigned short* hTb  = (unsigned short*)(ws + 13107200);   //  4,194,304 B
    float*          fsrc = (float*)         (ws + 17301504);   //     65,536 B
    float*          fdst = (float*)         (ws + 17367040);   //     65,536 B

    k_x2bf  <<<4096, 256, 0, stream>>>(x, xb, 4096 * 512);
    k_ham   <<<(4 * 128 * 512 + 255) / 256, 256, 0, stream>>>(W, hamT);
    k_gemm_h<<<dim3(32, 2, 4), 256, 0, stream>>>(xb, hamT, hf, hTb);
    k_logits<<<(4 * 4096 + 255) / 256, 256, 0, stream>>>(hf, a, fsrc, fdst);
    size_t smem = 16 * 4096 * 2 + 256 * 4 + 16 * 4;
    k_attn  <<<dim3(256, 4), 256, smem, stream>>>(adj, fsrc, fdst, hTb, out);
}
